// BlockAttentionResidual_35854386987551
// MI455X (gfx1250) — compile-verified, matrix operand order checked by reading, not run on hardware
//
#include <hip/hip_runtime.h>
#include <hip/hip_bf16.h>
#include <math.h>

// ---------------- problem constants ----------------
#define P      7
#define BATCH  4
#define SEQ    2048
#define D      1024
#define H      16
#define HD     64
#define NTOK   (BATCH*SEQ)        // 8192
#define TT     16                 // token tile (WMMA M)
#define NBLK   (NTOK/TT)          // 512 workgroups
#define SCALE  0.125f             // HD^-0.5

// ---------------- workspace layout (bytes) ----------------
#define WS_PWV   0                        // packed Wv  bf16 : 1024*1024*2 = 2 MB
#define WS_PWO   (2*1024*1024)            // packed Wo  bf16 : 2 MB
#define WS_PQF   (4*1024*1024)            // packed qfold bf16: 32 ktiles *32 lanes*16 = 32 KB
#define WS_SB    (4*1024*1024 + 32768)    // sbias: 16 floats

// ---------------- LDS layout (bytes) ----------------
#define L_X      0                        // 7 tiles * [16][1024] bf16 = 229376
#define L_XMIX   (7*32768)                // 2 tiles * [16][1024] bf16 =  65536
#define L_SC     (L_XMIX + 65536)         // scores/attn [7][16][16] f32 = 7168
#define L_AO     (L_SC + 7168)            // ao staging [16][128] bf16   = 4096
#define L_TOTAL  (L_AO + 4096)            // 306176 < 327680 (320 KB WGP LDS)

typedef __attribute__((ext_vector_type(16))) __bf16 v16bf;
typedef __attribute__((ext_vector_type(8)))  float  v8f;

union BF16x16 { v16bf v; uint4 q[2]; };
static_assert(sizeof(v16bf) == 32, "v16bf must be 32 bytes");

// fp32 -> bf16 (round-to-nearest-even), bf16 -> fp32
__device__ __forceinline__ unsigned f2bf(float f) {
    unsigned u = __builtin_bit_cast(unsigned, f);
    return (u + 0x7fffu + ((u >> 16) & 1u)) >> 16;
}
__device__ __forceinline__ float bf2f(unsigned b) {
    unsigned u = (b & 0xffffu) << 16;
    return __builtin_bit_cast(float, u);
}
__device__ __forceinline__ unsigned pack2(float a, float b) {
    return f2bf(a) | (f2bf(b) << 16);
}

// A-operand (16xK bf16) from LDS tile with XOR-by-row 16B-chunk swizzle.
// Per ISA 16-bit A layout: lane m=lane&15 holds row m; lanes 0-15 hold
// K=kt*32+{0..7,16..23}, lanes 16-31 hold K=kt*32+{8..15,24..31}.
__device__ __forceinline__ v16bf lds_load_a(const char* base, int row,
                                            int rowStrideB, int c1) {
    BF16x16 r;
    r.q[0] = *(const uint4*)(base + row*rowStrideB + (((c1    ) ^ row) << 4));
    r.q[1] = *(const uint4*)(base + row*rowStrideB + (((c1 + 2) ^ row) << 4));
    return r.v;
}

// B-operand from pre-packed global buffer: 32 contiguous bytes per lane.
__device__ __forceinline__ v16bf gl_load_b(const unsigned short* pk, int tile, int lane) {
    BF16x16 r;
    const uint4* p = (const uint4*)(pk + ((size_t)tile*32 + lane)*16);
    r.q[0] = p[0]; r.q[1] = p[1];
    return r.v;
}

// ---------------- prep: pack a [1024,1024] fp32 weight into WMMA-B bf16 layout ----
extern "C" __global__ void __launch_bounds__(256)
prep_packw(const float* __restrict__ W, unsigned short* __restrict__ pk) {
    int s = blockIdx.x*256 + threadIdx.x;       // 1,048,576 slots
    int e    = s & 15;
    int lane = (s >> 4) & 31;
    int tile = s >> 9;                          // kt*64 + nt
    int nt = tile & 63, kt = tile >> 6;
    int n  = nt*16 + (lane & 15);
    int base = (lane < 16) ? 0 : 8;
    int k  = kt*32 + ((e < 8) ? (base + e) : (16 + base + (e - 8)));
    pk[s] = (unsigned short)f2bf(W[(size_t)k*D + n]);
}

// ---------------- prep: fold q into Wk, pack into WMMA-B layout (N = 16 heads) ----
extern "C" __global__ void __launch_bounds__(256)
prep_qfold(const float* __restrict__ Wk, const float* __restrict__ bk,
           const float* __restrict__ pq, const int* __restrict__ bidx,
           unsigned short* __restrict__ pQf, float* __restrict__ sbias) {
    int s = blockIdx.x*256 + threadIdx.x;       // 16,384 slots (32 ktiles, 1 ntile)
    const float* q = pq + (size_t)bidx[0]*H*HD;
    int e    = s & 15;
    int lane = (s >> 4) & 31;
    int kt   = s >> 9;
    int h    = lane & 15;
    int base = (lane < 16) ? 0 : 8;
    int k  = kt*32 + ((e < 8) ? (base + e) : (16 + base + (e - 8)));
    float acc = 0.f;
    const float* wrow = Wk + (size_t)k*D + h*HD;
    const float* qrow = q + h*HD;
    #pragma unroll 8
    for (int j = 0; j < HD; ++j) acc += wrow[j]*qrow[j];
    pQf[s] = (unsigned short)f2bf(acc * SCALE);
    if (s < H) {
        float b = 0.f;
        for (int j = 0; j < HD; ++j) b += bk[s*HD + j]*q[s*HD + j];
        sbias[s] = b * SCALE;
    }
}

// ---------------- fused main kernel: scores -> softmax(P) -> mix -> Wv -> Wo ------
extern "C" __global__ void __launch_bounds__(256)
fused_attn_kernel(const float* __restrict__ x,          // [P][NTOK][D] fp32
                  const float* __restrict__ bv,
                  const float* __restrict__ bo,
                  const unsigned short* __restrict__ pWv,
                  const unsigned short* __restrict__ pWo,
                  const unsigned short* __restrict__ pQf,
                  const float* __restrict__ sbias,
                  float* __restrict__ out) {
    extern __shared__ char smem[];
    const int tid  = threadIdx.x;
    const int lane = tid & 31;
    const int wave = tid >> 5;
    const int hi16 = lane >> 4;
    const int m    = lane & 15;
    const int tokBase = blockIdx.x * TT;

    // ---- Stage 1: stream x tiles (7 x [16][1024] fp32) -> bf16 LDS, swizzled ----
    #pragma unroll 2
    for (int i = 0; i < 56; ++i) {
        int cid = tid + 256*i;                 // 14336 16-byte chunks
        int row = cid >> 7;                    // p*16 + t
        int c   = cid & 127;
        int p = row >> 4, t = row & 15;
        const float4* src = (const float4*)(x + ((size_t)p*NTOK + tokBase + t)*D + c*8);
        float4 a = src[0], b = src[1];
        uint4 v;
        v.x = pack2(a.x, a.y); v.y = pack2(a.z, a.w);
        v.z = pack2(b.x, b.y); v.w = pack2(b.z, b.w);
        *(uint4*)(smem + L_X + p*32768 + t*2048 + ((c ^ t) << 4)) = v;
    }
    __syncthreads();

    // ---- Stage 2: scores[p][t][h] = x_p @ qfold  (one wave per p, WMMA) ----
    if (wave < P) {
        const int p = wave;
        const char* xb = smem + L_X + p*32768;
        v8f s = {0.f,0.f,0.f,0.f,0.f,0.f,0.f,0.f};
        #pragma unroll 4
        for (int kt = 0; kt < 32; ++kt) {
            v16bf a = lds_load_a(xb, m, 2048, kt*4 + hi16);
            v16bf b = gl_load_b(pQf, kt, lane);
            s = __builtin_amdgcn_wmma_f32_16x16x32_bf16(false, a, false, b,
                                                        (short)0, s, false, false);
        }
        float sb = sbias[m];                   // head = D-lane column
        float* sc = (float*)(smem + L_SC);
        #pragma unroll
        for (int r = 0; r < 8; ++r) {
            int row = r + 8*hi16;              // token
            sc[p*256 + row*16 + m] = s[r] + sb;
        }
    }
    __syncthreads();

    // ---- Stage 3: softmax over P, in place (256 threads == 16t x 16h) ----
    {
        float* sc = (float*)(smem + L_SC);
        int t = tid >> 4, h = tid & 15;
        float v[P]; float mx = -1e30f;
        #pragma unroll
        for (int p = 0; p < P; ++p) { v[p] = sc[p*256 + t*16 + h]; mx = fmaxf(mx, v[p]); }
        float sum = 0.f;
        #pragma unroll
        for (int p = 0; p < P; ++p) { v[p] = __expf(v[p] - mx); sum += v[p]; }
        float inv = 1.0f / sum;
        #pragma unroll
        for (int p = 0; p < P; ++p) sc[p*256 + t*16 + h] = v[p]*inv;
    }
    __syncthreads();

    // ---- Stage 4: per head-pair: xmix -> (xmix@Wv_h + bv) -> rank-128 @Wo ----
    v8f oacc[8];
    #pragma unroll
    for (int i = 0; i < 8; ++i) oacc[i] = (v8f){0.f,0.f,0.f,0.f,0.f,0.f,0.f,0.f};
    const float* sc = (const float*)(smem + L_SC);

    for (int hp = 0; hp < 8; ++hp) {
        // build xmix for head 2hp (tids 0-127) and 2hp+1 (tids 128-255)
        {
            int half = tid >> 7;
            int u = tid & 127;
            int t = u & 15;                    // token (conflict-free across g)
            int g = u >> 4;                    // chunk phase 0..7
            int h = hp*2 + half;
            float a[P];
            #pragma unroll
            for (int p = 0; p < P; ++p) a[p] = sc[p*256 + t*16 + h];
            char* dst = smem + L_XMIX + half*32768;
            for (int j = 0; j < 16; ++j) {
                int c = g + 8*j;
                float acc[8] = {0,0,0,0,0,0,0,0};
                #pragma unroll
                for (int p = 0; p < P; ++p) {
                    uint4 xv = *(const uint4*)(smem + L_X + p*32768 + t*2048 + ((c ^ t) << 4));
                    float ap = a[p];
                    acc[0] += ap*bf2f(xv.x);  acc[1] += ap*bf2f(xv.x >> 16);
                    acc[2] += ap*bf2f(xv.y);  acc[3] += ap*bf2f(xv.y >> 16);
                    acc[4] += ap*bf2f(xv.z);  acc[5] += ap*bf2f(xv.z >> 16);
                    acc[6] += ap*bf2f(xv.w);  acc[7] += ap*bf2f(xv.w >> 16);
                }
                uint4 o;
                o.x = pack2(acc[0], acc[1]); o.y = pack2(acc[2], acc[3]);
                o.z = pack2(acc[4], acc[5]); o.w = pack2(acc[6], acc[7]);
                *(uint4*)(dst + t*2048 + ((c ^ t) << 4)) = o;
            }
        }
        __syncthreads();

        // V GEMM: waves 0-3 -> head 2hp (ntile=wave), waves 4-7 -> head 2hp+1
        {
            int hsel = wave >> 2;
            int h    = hp*2 + hsel;
            int nt   = wave & 3;
            const char* ab = smem + L_XMIX + hsel*32768;
            int ntile = h*4 + nt;
            v8f c8 = {0.f,0.f,0.f,0.f,0.f,0.f,0.f,0.f};
            #pragma unroll 4
            for (int kt = 0; kt < 32; ++kt) {
                v16bf a = lds_load_a(ab, m, 2048, kt*4 + hi16);
                v16bf b = gl_load_b(pWv, kt*64 + ntile, lane);
                c8 = __builtin_amdgcn_wmma_f32_16x16x32_bf16(false, a, false, b,
                                                             (short)0, c8, false, false);
            }
            float bvv = bv[h*64 + nt*16 + m];
            int aocol = hsel*64 + nt*16 + m;   // column in [16][128] ao tile
            int chunk = aocol >> 3;
            int off   = (aocol & 7) << 1;
            #pragma unroll
            for (int r = 0; r < 8; ++r) {
                int row = r + 8*hi16;
                *(unsigned short*)(smem + L_AO + row*256 + ((chunk ^ row) << 4) + off)
                    = (unsigned short)f2bf(c8[r] + bvv);
            }
        }
        __syncthreads();

        // rank-128 Wo update into persistent accumulators (wave owns 128 cols)
        #pragma unroll
        for (int nt2 = 0; nt2 < 8; ++nt2) {
            int ntg = wave*8 + nt2;
            v8f c8 = oacc[nt2];
            #pragma unroll
            for (int kt = 0; kt < 4; ++kt) {
                v16bf a = lds_load_a(smem + L_AO, m, 256, kt*4 + hi16);
                v16bf b = gl_load_b(pWo, (hp*4 + kt)*64 + ntg, lane);
                c8 = __builtin_amdgcn_wmma_f32_16x16x32_bf16(false, a, false, b,
                                                             (short)0, c8, false, false);
            }
            oacc[nt2] = c8;
        }
        // no barrier needed: next iter's xmix/ao writes are fenced by the two
        // barriers above before any wave can touch them again.
    }

    // ---- epilogue: += bo, store fp32 ----
    #pragma unroll
    for (int nt2 = 0; nt2 < 8; ++nt2) {
        int col = wave*128 + nt2*16 + m;
        float bov = bo[col];
        #pragma unroll
        for (int r = 0; r < 8; ++r) {
            int row = r + 8*hi16;
            out[(size_t)(tokBase + row)*D + col] = oacc[nt2][r] + bov;
        }
    }
}

// ---------------- host launch ----------------
extern "C" void kernel_launch(void* const* d_in, const int* in_sizes, int n_in,
                              void* d_out, int out_size, void* d_ws, size_t ws_size,
                              hipStream_t stream) {
    (void)in_sizes; (void)n_in; (void)out_size; (void)ws_size;
    const float* prev = (const float*)d_in[0];
    const float* pq   = (const float*)d_in[1];
    const float* Wk   = (const float*)d_in[2];
    const float* bk   = (const float*)d_in[3];
    const float* Wv   = (const float*)d_in[4];
    const float* bv   = (const float*)d_in[5];
    const float* Wo   = (const float*)d_in[6];
    const float* bo   = (const float*)d_in[7];
    const int*   bidx = (const int*)d_in[8];
    float* out = (float*)d_out;
    char*  ws  = (char*)d_ws;

    unsigned short* pWv = (unsigned short*)(ws + WS_PWV);
    unsigned short* pWo = (unsigned short*)(ws + WS_PWO);
    unsigned short* pQf = (unsigned short*)(ws + WS_PQF);
    float*          sb  = (float*)(ws + WS_SB);

    prep_packw<<<4096, 256, 0, stream>>>(Wv, pWv);
    prep_packw<<<4096, 256, 0, stream>>>(Wo, pWo);
    prep_qfold<<<64,   256, 0, stream>>>(Wk, bk, pq, bidx, pQf, sb);
    fused_attn_kernel<<<NBLK, 256, L_TOTAL, stream>>>(prev, bv, bo, pWv, pWo, pQf, sb, out);
}
